// MultiHeadAttentionBlock_60249801228819
// MI455X (gfx1250) — matrix operand order repaired and checked by reading, not compiled, not run on hardware
//
#include <hip/hip_runtime.h>
#include <hip/hip_bf16.h>
#include <math.h>

// ---------------------------------------------------------------------------
// MHA block for gfx1250 (MI455X): f16 WMMA (16x16x32), f32 accumulate.
// Pre-pass converts all f32 operands to f16 once; hot loops are pure
// b128 fragment loads + v_wmma.  Attention uses a 32-query tile per wave so
// each K/V fragment load feeds 2 WMMAs (32 FLOP/byte from L2).
// ---------------------------------------------------------------------------

typedef _Float16 half_t;
typedef __attribute__((ext_vector_type(16))) _Float16     v16h;
typedef __attribute__((ext_vector_type(8)))  _Float16     v8h;
typedef __attribute__((ext_vector_type(8)))  float        v8f;
typedef __attribute__((ext_vector_type(4)))  float        v4f;
typedef __attribute__((ext_vector_type(4)))  unsigned int v4u;

union FragU { v16h v; v4u q[2]; };   // 32 bytes, both members trivial

#define WMMA_F16(a, b, c) \
  __builtin_amdgcn_wmma_f32_16x16x32_f16(false, (a), false, (b), (short)0, (c), false, false)

static constexpr int BB = 2, SS = 2048, DD = 1024, HH = 16, DKK = 64;

static __device__ inline v8f vzero8() {
  v8f z;
#pragma unroll
  for (int i = 0; i < 8; ++i) z[i] = 0.f;
  return z;
}

// ---------------------------------------------------------------------------
// Kernel 0: elementwise f32 -> f16 (vectorized 8 elems/thread)
// ---------------------------------------------------------------------------
__global__ void cvt_f32_to_f16(const float* __restrict__ src,
                               half_t* __restrict__ dst, int n) {
  const int i = (blockIdx.x * blockDim.x + threadIdx.x) * 8;
  if (i >= n) return;
  const v4f x0 = *(const v4f*)(src + i);
  const v4f x1 = *(const v4f*)(src + i + 4);
  v8h h;
#pragma unroll
  for (int j = 0; j < 4; ++j) { h[j] = (_Float16)x0[j]; h[4 + j] = (_Float16)x1[j]; }
  *(v8h*)(dst + i) = h;
}

// ---------------------------------------------------------------------------
// Kernel 1: fused Q/K/V projections, pure f16 GEMM.
// C[m,n] = act[m,:] . W[n,:] + bias[n]
// grid = (M/32, N/64, 3); block = 32 (one wave computes a 32x64 tile).
// Q,K stored f16 [B,H,S,DK]; V stored f16 transposed [B,H,DK,S].
// ---------------------------------------------------------------------------
__global__ void mha_proj_qkv(const half_t* __restrict__ qh,
                             const half_t* __restrict__ kh,
                             const half_t* __restrict__ vh,
                             const half_t* __restrict__ wqh,
                             const half_t* __restrict__ wkh,
                             const half_t* __restrict__ wvh,
                             const float* __restrict__ bq,
                             const float* __restrict__ bk,
                             const float* __restrict__ bv,
                             half_t* __restrict__ Qh, half_t* __restrict__ Kh,
                             half_t* __restrict__ Vt) {
  const int lane = threadIdx.x & 31;
  const int lrow = lane & 15;
  const int hi   = lane >> 4;
  const int mtile  = blockIdx.x;    // 32 rows of B*S
  const int ntile4 = blockIdx.y;    // 64 cols of D
  const int which  = blockIdx.z;    // 0=Q 1=K 2=V

  const half_t* act;  const half_t* W;  const float* bias;
  if (which == 0)      { act = qh; W = wqh; bias = bq; }
  else if (which == 1) { act = kh; W = wkh; bias = bk; }
  else                 { act = vh; W = wvh; bias = bv; }

  const half_t* arow0 = act + (size_t)(mtile * 32 + lrow) * DD;
  const half_t* arow1 = act + (size_t)(mtile * 32 + 16 + lrow) * DD;

  v8f acc[2][4];
#pragma unroll
  for (int i = 0; i < 2; ++i)
#pragma unroll
    for (int t = 0; t < 4; ++t) acc[i][t] = vzero8();

  for (int kk = 0; kk < DD; kk += 32) {
    // issue ALL fragment loads first (independent regs -> loads stay in flight)
    FragU a0, a1, bf[4];
    a0.q[0] = *(const v4u*)(arow0 + kk + hi * 8);
    a0.q[1] = *(const v4u*)(arow0 + kk + 16 + hi * 8);
    a1.q[0] = *(const v4u*)(arow1 + kk + hi * 8);
    a1.q[1] = *(const v4u*)(arow1 + kk + 16 + hi * 8);
#pragma unroll
    for (int nt = 0; nt < 4; ++nt) {
      const half_t* wrow = W + (size_t)(ntile4 * 64 + nt * 16 + lrow) * DD + kk + hi * 8;
      bf[nt].q[0] = *(const v4u*)wrow;
      bf[nt].q[1] = *(const v4u*)(wrow + 16);
    }
    if (kk + 32 < DD) {                        // WGP-scope prefetch of next A panel
      __builtin_prefetch(arow0 + kk + 32, 0, 3);
      __builtin_prefetch(arow1 + kk + 32, 0, 3);
    }
#pragma unroll
    for (int nt = 0; nt < 4; ++nt) {
      acc[0][nt] = WMMA_F16(a0.v, bf[nt].v, acc[0][nt]);
      acc[1][nt] = WMMA_F16(a1.v, bf[nt].v, acc[1][nt]);
    }
  }

  // Epilogue: bias + scatter to head layout
#pragma unroll
  for (int nt = 0; nt < 4; ++nt) {
    const int n  = ntile4 * 64 + nt * 16 + lrow;
    const float bn = bias[n];
    const int h  = n >> 6;
    const int dk = n & 63;
#pragma unroll
    for (int i = 0; i < 2; ++i) {
#pragma unroll
      for (int r = 0; r < 8; ++r) {
        const int mr = mtile * 32 + i * 16 + r + hi * 8;   // C layout: VGPR r -> row r / r+8
        const int b  = mr >> 11;
        const int s  = mr & 2047;
        const float val = acc[i][nt][r] + bn;
        if (which == 2) {
          Vt[(((size_t)b * HH + h) * DKK + dk) * SS + s] = (_Float16)val;
        } else {
          half_t* dst = (which == 0) ? Qh : Kh;
          dst[(((size_t)b * HH + h) * SS + s) * DKK + dk] = (_Float16)val;
        }
      }
    }
  }
}

// ---------------------------------------------------------------------------
// Kernel 2: flash attention.  One wave per (head, 32-query tile).
// grid = (S/32, B*H); block = 32.  K/V fragments are shared by the two
// 16-row groups -> 16 WMMAs per 16 b128 loads per key step.
// ---------------------------------------------------------------------------
__global__ void mha_attention(const half_t* __restrict__ Qh,
                              const half_t* __restrict__ Kh,
                              const half_t* __restrict__ Vt,
                              const int* __restrict__ mask,
                              half_t* __restrict__ X) {
  __shared__ alignas(16) _Float16 lds_p[32 * 32];   // P tile staging (C->A transpose)

  const int lane = threadIdx.x & 31;
  const int lrow = lane & 15;
  const int hi   = lane >> 4;
  const int qtile = blockIdx.x;           // 0..63, 32 query rows each
  const int bh    = blockIdx.y;
  const int b = bh >> 4, h = bh & 15;

  const half_t* Qp = Qh + (size_t)bh * SS * DKK;
  const half_t* Kp = Kh + (size_t)bh * SS * DKK;
  const half_t* Vp = Vt + (size_t)bh * DKK * SS;

  // Q fragments, resident: a[group][dk-block]
  FragU a[2][2];
#pragma unroll
  for (int i = 0; i < 2; ++i) {
    const half_t* qr = Qp + (size_t)(qtile * 32 + i * 16 + lrow) * DKK;
    a[i][0].q[0] = *(const v4u*)(qr + hi * 8);
    a[i][0].q[1] = *(const v4u*)(qr + 16 + hi * 8);
    a[i][1].q[0] = *(const v4u*)(qr + 32 + hi * 8);
    a[i][1].q[1] = *(const v4u*)(qr + 48 + hi * 8);
  }

  v8f o[2][4];
#pragma unroll
  for (int i = 0; i < 2; ++i)
#pragma unroll
    for (int t = 0; t < 4; ++t) o[i][t] = vzero8();
  float mstat[2][8], lstat[2][8];
#pragma unroll
  for (int i = 0; i < 2; ++i)
#pragma unroll
    for (int r = 0; r < 8; ++r) { mstat[i][r] = -INFINITY; lstat[i][r] = 0.f; }

  const float scale = 0.125f;   // 1/sqrt(DK)

  for (int k0 = 0; k0 < SS; k0 += 32) {
    // ---- K fragments (shared by both row groups): bk[key-ntile][dk-block]
    const half_t* kr0 = Kp + (size_t)(k0 + lrow) * DKK;
    const half_t* kr1 = Kp + (size_t)(k0 + 16 + lrow) * DKK;
    FragU bk[2][2];
    bk[0][0].q[0] = *(const v4u*)(kr0 + hi * 8);
    bk[0][0].q[1] = *(const v4u*)(kr0 + hi * 8 + 16);
    bk[0][1].q[0] = *(const v4u*)(kr0 + 32 + hi * 8);
    bk[0][1].q[1] = *(const v4u*)(kr0 + 32 + hi * 8 + 16);
    bk[1][0].q[0] = *(const v4u*)(kr1 + hi * 8);
    bk[1][0].q[1] = *(const v4u*)(kr1 + hi * 8 + 16);
    bk[1][1].q[0] = *(const v4u*)(kr1 + 32 + hi * 8);
    bk[1][1].q[1] = *(const v4u*)(kr1 + 32 + hi * 8 + 16);

    // ---- per row group: scores + online softmax + stage P into LDS
#pragma unroll
    for (int i = 0; i < 2; ++i) {
      v8f s0 = vzero8(), s1 = vzero8();
      s0 = WMMA_F16(a[i][0].v, bk[0][0].v, s0);
      s0 = WMMA_F16(a[i][1].v, bk[0][1].v, s0);
      s1 = WMMA_F16(a[i][0].v, bk[1][0].v, s1);
      s1 = WMMA_F16(a[i][1].v, bk[1][1].v, s1);

#pragma unroll
      for (int r = 0; r < 8; ++r) {
        const int qg = qtile * 32 + i * 16 + r + hi * 8;
        const int kg0 = k0 + lrow, kg1 = k0 + 16 + lrow;
        float v0 = s0[r] * scale;
        float v1 = s1[r] * scale;
        v0 = (mask[(size_t)qg * SS + kg0] == 0) ? -1e9f : v0;
        v1 = (mask[(size_t)qg * SS + kg1] == 0) ? -1e9f : v1;

        float tmax = fmaxf(v0, v1);
#pragma unroll
        for (int off = 8; off >= 1; off >>= 1)   // reduce within 16-lane group
          tmax = fmaxf(tmax, __shfl_xor(tmax, off, 32));
        const float mnew = fmaxf(mstat[i][r], tmax);
        const float p0 = __expf(v0 - mnew);
        const float p1 = __expf(v1 - mnew);
        float psum = p0 + p1;
#pragma unroll
        for (int off = 8; off >= 1; off >>= 1)
          psum += __shfl_xor(psum, off, 32);
        const float corr = __expf(mstat[i][r] - mnew);
        lstat[i][r] = lstat[i][r] * corr + psum;
        mstat[i][r] = mnew;
#pragma unroll
        for (int t = 0; t < 4; ++t) o[i][t][r] *= corr;   // rescale running output

        lds_p[(i * 16 + r + hi * 8) * 32 + lrow]      = (_Float16)p0;
        lds_p[(i * 16 + r + hi * 8) * 32 + 16 + lrow] = (_Float16)p1;
      }
    }
    __syncthreads();   // single-wave WG: cheap; orders DS stores -> DS loads

    // read P as A fragments: row = group*16 + lane%16
    FragU pf[2];
#pragma unroll
    for (int i = 0; i < 2; ++i) {
      const _Float16* pr = lds_p + (i * 16 + lrow) * 32;
      pf[i].q[0] = *(const v4u*)(pr + hi * 8);
      pf[i].q[1] = *(const v4u*)(pr + 16 + hi * 8);
    }
    __syncthreads();   // protect lds_p before next iteration rewrites it

    // ---- V fragments (shared): O[32x64] += P[32x32] . V[32x64]
    FragU bv[4];
#pragma unroll
    for (int nt = 0; nt < 4; ++nt) {
      const half_t* vr = Vp + (size_t)(nt * 16 + lrow) * SS + k0 + hi * 8;
      bv[nt].q[0] = *(const v4u*)vr;
      bv[nt].q[1] = *(const v4u*)(vr + 16);
    }
#pragma unroll
    for (int nt = 0; nt < 4; ++nt) {
      o[0][nt] = WMMA_F16(pf[0].v, bv[nt].v, o[0][nt]);
      o[1][nt] = WMMA_F16(pf[1].v, bv[nt].v, o[1][nt]);
    }
  }

  // ---- normalize and write X[B,S,D] (f16) with heads re-interleaved
#pragma unroll
  for (int i = 0; i < 2; ++i) {
#pragma unroll
    for (int r = 0; r < 8; ++r) {
      const float inv = 1.0f / lstat[i][r];
      const int qg = qtile * 32 + i * 16 + r + hi * 8;
#pragma unroll
      for (int nt = 0; nt < 4; ++nt) {
        const int dcol = h * DKK + nt * 16 + lrow;
        X[((size_t)(b * SS + qg)) * DD + dcol] = (_Float16)(o[i][nt][r] * inv);
      }
    }
  }
}

// ---------------------------------------------------------------------------
// Kernel 3: output projection (pure f16 GEMM, f32 output + bias).
// grid = (M/32, N/64); block = 32.
// ---------------------------------------------------------------------------
__global__ void mha_out_proj(const half_t* __restrict__ X,
                             const half_t* __restrict__ woh,
                             const float* __restrict__ bo,
                             float* __restrict__ out) {
  const int lane = threadIdx.x & 31;
  const int lrow = lane & 15;
  const int hi   = lane >> 4;
  const int mtile  = blockIdx.x;
  const int ntile4 = blockIdx.y;

  const half_t* xr0 = X + (size_t)(mtile * 32 + lrow) * DD;
  const half_t* xr1 = X + (size_t)(mtile * 32 + 16 + lrow) * DD;

  v8f acc[2][4];
#pragma unroll
  for (int i = 0; i < 2; ++i)
#pragma unroll
    for (int t = 0; t < 4; ++t) acc[i][t] = vzero8();

  for (int kk = 0; kk < DD; kk += 32) {
    FragU a0, a1, bf[4];
    a0.q[0] = *(const v4u*)(xr0 + kk + hi * 8);
    a0.q[1] = *(const v4u*)(xr0 + kk + 16 + hi * 8);
    a1.q[0] = *(const v4u*)(xr1 + kk + hi * 8);
    a1.q[1] = *(const v4u*)(xr1 + kk + 16 + hi * 8);
#pragma unroll
    for (int nt = 0; nt < 4; ++nt) {
      const half_t* wrow = woh + (size_t)(ntile4 * 64 + nt * 16 + lrow) * DD + kk + hi * 8;
      bf[nt].q[0] = *(const v4u*)wrow;
      bf[nt].q[1] = *(const v4u*)(wrow + 16);
    }
    if (kk + 32 < DD) {
      __builtin_prefetch(xr0 + kk + 32, 0, 3);
      __builtin_prefetch(xr1 + kk + 32, 0, 3);
    }
#pragma unroll
    for (int nt = 0; nt < 4; ++nt) {
      acc[0][nt] = WMMA_F16(a0.v, bf[nt].v, acc[0][nt]);
      acc[1][nt] = WMMA_F16(a1.v, bf[nt].v, acc[1][nt]);
    }
  }

#pragma unroll
  for (int nt = 0; nt < 4; ++nt) {
    const int n  = ntile4 * 64 + nt * 16 + lrow;
    const float bn = bo[n];
#pragma unroll
    for (int i = 0; i < 2; ++i) {
#pragma unroll
      for (int r = 0; r < 8; ++r) {
        const int mr = mtile * 32 + i * 16 + r + hi * 8;
        out[(size_t)mr * DD + n] = acc[i][nt][r] + bn;
      }
    }
  }
}

// ---------------------------------------------------------------------------
extern "C" void kernel_launch(void* const* d_in, const int* in_sizes, int n_in,
                              void* d_out, int out_size, void* d_ws, size_t ws_size,
                              hipStream_t stream) {
  const float* q    = (const float*)d_in[0];
  const float* k    = (const float*)d_in[1];
  const float* v    = (const float*)d_in[2];
  const int*   mask = (const int*)  d_in[3];
  const float* w_q  = (const float*)d_in[4];
  const float* b_q  = (const float*)d_in[5];
  const float* w_k  = (const float*)d_in[6];
  const float* b_k  = (const float*)d_in[7];
  const float* w_v  = (const float*)d_in[8];
  const float* b_v  = (const float*)d_in[9];
  const float* w_o  = (const float*)d_in[10];
  const float* b_o  = (const float*)d_in[11];
  float* out = (float*)d_out;

  // workspace layout (all f16):
  //   qh|kh|vh          : B*S*D each      (3 x 8 MB)
  //   wqh|wkh|wvh|woh   : D*D each        (4 x 2 MB)
  //   Qh|Kh|Vt|Xh       : B*H*S*DK each   (4 x 8 MB)
  const size_t nAct = (size_t)BB * SS * DD;   // 4 Mi elems
  const size_t nW   = (size_t)DD * DD;        // 1 Mi elems
  half_t* qh  = (half_t*)d_ws;
  half_t* kh  = qh  + nAct;
  half_t* vh  = kh  + nAct;
  half_t* wqh = vh  + nAct;
  half_t* wkh = wqh + nW;
  half_t* wvh = wkh + nW;
  half_t* wvh2= wvh;  (void)wvh2;
  half_t* woh = wvh + nW;
  half_t* Qh  = woh + nW;
  half_t* Kh  = Qh  + nAct;
  half_t* Vt  = Kh  + nAct;
  half_t* Xh  = Vt  + nAct;

  const dim3 blk(32);
  const int ct = 256;
  cvt_f32_to_f16<<<dim3((int)(nAct / 8 / ct)), ct, 0, stream>>>(q, qh, (int)nAct);
  cvt_f32_to_f16<<<dim3((int)(nAct / 8 / ct)), ct, 0, stream>>>(k, kh, (int)nAct);
  cvt_f32_to_f16<<<dim3((int)(nAct / 8 / ct)), ct, 0, stream>>>(v, vh, (int)nAct);
  cvt_f32_to_f16<<<dim3((int)(nW / 8 / ct)),   ct, 0, stream>>>(w_q, wqh, (int)nW);
  cvt_f32_to_f16<<<dim3((int)(nW / 8 / ct)),   ct, 0, stream>>>(w_k, wkh, (int)nW);
  cvt_f32_to_f16<<<dim3((int)(nW / 8 / ct)),   ct, 0, stream>>>(w_v, wvh, (int)nW);
  cvt_f32_to_f16<<<dim3((int)(nW / 8 / ct)),   ct, 0, stream>>>(w_o, woh, (int)nW);

  mha_proj_qkv<<<dim3((BB * SS) / 32, DD / 64, 3), blk, 0, stream>>>(
      qh, kh, vh, wqh, wkh, wvh, b_q, b_k, b_v, Qh, Kh, Vt);
  mha_attention<<<dim3(SS / 32, BB * HH), blk, 0, stream>>>(Qh, Kh, Vt, mask, Xh);
  mha_out_proj<<<dim3((BB * SS) / 32, DD / 64), blk, 0, stream>>>(Xh, woh, b_o, out);
}
